// KMeans_17772574671263
// MI455X (gfx1250) — matrix operand order repaired and checked by reading, not compiled, not run on hardware
//
#include <hip/hip_runtime.h>

#define N_PTS   131072
#define DIM     256
#define NCLUST  1024
#define NITERS  10

typedef __attribute__((ext_vector_type(16))) __bf16 v16bf;
typedef __attribute__((ext_vector_type(8)))  float  v8f;

__device__ __forceinline__ unsigned f32_bits(float f) {
    union { float f; unsigned u; } v; v.f = f; return v.u;
}
__device__ __forceinline__ float bits_f32(unsigned u) {
    union { float f; unsigned u; } v; v.u = u; return v.f;
}
// Split an fp32 into bf16 hi (truncation); also return hi as fp32.
__device__ __forceinline__ __bf16 bf_hi(float f, float& hif) {
    unsigned u = f32_bits(f) & 0xFFFF0000u;
    hif = bits_f32(u);
    unsigned short s = (unsigned short)(u >> 16);
    return __builtin_bit_cast(__bf16, s);
}
__device__ __forceinline__ __bf16 bf_trunc(float f) {
    unsigned short s = (unsigned short)(f32_bits(f) >> 16);
    return __builtin_bit_cast(__bf16, s);
}

// ---------------------------------------------------------------- init: centroids = x[:K]
__global__ void __launch_bounds__(256)
k_init_centroids(const float* __restrict__ x, float* __restrict__ centroids) {
    int i = blockIdx.x * 256 + threadIdx.x;           // K*D = 262144 elements
    centroids[i] = x[i];
}

// ---------------------------------------------------------------- per-iter: split centroids to bf16 hi/lo, compute c2
__global__ void __launch_bounds__(256)
k_prep_centroids(const float* __restrict__ centroids,
                 __bf16* __restrict__ chg, __bf16* __restrict__ clg,
                 float* __restrict__ c2) {
    __shared__ float red[256];
    int k = blockIdx.x, t = threadIdx.x;
    float v = centroids[k * DIM + t];
    float hf;
    __bf16 h = bf_hi(v, hf);
    chg[k * DIM + t] = h;
    clg[k * DIM + t] = bf_trunc(v - hf);
    red[t] = v * v;
    __syncthreads();
    for (int s = 128; s > 0; s >>= 1) {
        if (t < s) red[t] += red[t + s];
        __syncthreads();
    }
    if (t == 0) c2[k] = red[0];
}

// ---------------------------------------------------------------- zero sums + counts
__global__ void __launch_bounds__(256)
k_zero(float* __restrict__ sums, float* __restrict__ counts) {
    int i = blockIdx.x * 256 + threadIdx.x;
    if (i < NCLUST * DIM) sums[i] = 0.0f;
    if (i < NCLUST) counts[i] = 0.0f;
}

// ---------------------------------------------------------------- assignment: WMMA GEMM + fused argmin
// block = 256 threads = 8 waves; each wave owns 16 rows; block covers 128 rows.
// Centroid tiles (B) are double-buffer staged to LDS with async global->LDS copies.
__global__ void __launch_bounds__(256)
k_assign(const float* __restrict__ x,
         const __bf16* __restrict__ chg, const __bf16* __restrict__ clg,
         const float* __restrict__ c2, int* __restrict__ clusters) {
    // [parity][ hi: elems 0..4095 | lo: elems 4096..8191 ]  -> 2 x 16 KB
    __shared__ __align__(16) __bf16 sbuf[2][8192];

    const int t       = threadIdx.x;
    const int lane    = t & 31;
    const int wave    = t >> 5;
    const int rowBase = blockIdx.x * 128 + wave * 16;
    const int row     = rowBase + (lane & 15);
    const int kh      = (lane >> 4) * 16;       // A layout: lane half selects K 0-15 / 16-31
    const float* xr   = x + (size_t)row * DIM;

    // Load this wave's 16x256 A tile once, split fp32 -> bf16 hi/lo in registers.
    v16bf ahi[8], alo[8];
#pragma unroll
    for (int c = 0; c < 8; ++c) {
        const int db = c * 32;
        v8f f0 = *(const v8f*)(xr + db + kh);
        v8f f1 = *(const v8f*)(xr + db + kh + 8);
#pragma unroll
        for (int j = 0; j < 8; ++j) {
            float hf;
            ahi[c][j]     = bf_hi(f0[j], hf);
            alo[c][j]     = bf_trunc(f0[j] - hf);
            ahi[c][j + 8] = bf_hi(f1[j], hf);
            alo[c][j + 8] = bf_trunc(f1[j] - hf);
        }
    }

    float best[8];
    int   bidx[8];
#pragma unroll
    for (int i = 0; i < 8; ++i) { best[i] = 3.0e38f; bidx[i] = 0; }

    const int c16  = lane & 15;                 // column within the 16-wide tile
    const int kofs = (lane >> 4) * 16;          // B layout: lane half selects K 0-15 / 16-31
    const unsigned ldsbase = (unsigned)(unsigned long long)(void*)&sbuf[0][0];

    // Async-stage one 16-centroid tile (8KB hi + 8KB lo, contiguous) into LDS buffer `p`.
    auto stage = [&](int ct, int p) {
        unsigned ldst = ldsbase + (unsigned)p * 16384u + (unsigned)t * 16u;
        unsigned long long ghi =
            (unsigned long long)(const void*)(chg + (size_t)ct * 4096) + (unsigned long long)t * 16u;
        unsigned long long glo =
            (unsigned long long)(const void*)(clg + (size_t)ct * 4096) + (unsigned long long)t * 16u;
        asm volatile("global_load_async_to_lds_b128 %0, %1, off"
                     :: "v"(ldst), "v"(ghi) : "memory");
        asm volatile("global_load_async_to_lds_b128 %0, %1, off offset:4096"
                     :: "v"(ldst), "v"(ghi) : "memory");
        asm volatile("global_load_async_to_lds_b128 %0, %1, off"
                     :: "v"(ldst + 8192u), "v"(glo) : "memory");
        asm volatile("global_load_async_to_lds_b128 %0, %1, off offset:4096"
                     :: "v"(ldst + 8192u), "v"(glo) : "memory");
    };

    stage(0, 0);                                // prologue: tile 0 -> buf 0

    for (int ct = 0; ct < NCLUST / 16; ++ct) {
        const int p = ct & 1;
        if (ct < NCLUST / 16 - 1) {
            stage(ct + 1, p ^ 1);               // overlap next tile's copy with compute
            asm volatile("s_wait_asynccnt 4" ::: "memory");   // tile ct landed (in-order)
        } else {
            asm volatile("s_wait_asynccnt 0" ::: "memory");
        }
        __syncthreads();                        // all waves' async data visible

        const __bf16* bufp = &sbuf[p][0];
        v8f acc = {};
#pragma unroll
        for (int c = 0; c < 8; ++c) {
            const int db = c * 32;
            v16bf bh = *(const v16bf*)(bufp + c16 * DIM + db + kofs);          // ds_load
            v16bf bl = *(const v16bf*)(bufp + 4096 + c16 * DIM + db + kofs);   // ds_load
            acc = __builtin_amdgcn_wmma_f32_16x16x32_bf16(false, ahi[c], false, bh,
                                                          (short)0, acc, false, false);
            acc = __builtin_amdgcn_wmma_f32_16x16x32_bf16(false, ahi[c], false, bl,
                                                          (short)0, acc, false, false);
            acc = __builtin_amdgcn_wmma_f32_16x16x32_bf16(false, alo[c], false, bh,
                                                          (short)0, acc, false, false);
        }
        const int col   = ct * 16 + c16;        // centroid index this lane covers
        const float cc2 = c2[col];
#pragma unroll
        for (int i = 0; i < 8; ++i) {
            float score = cc2 - 2.0f * acc[i];  // argmin(d2) == argmin(c2 - 2 x.c)
            bool take = score < best[i];
            best[i] = take ? score : best[i];
            bidx[i] = take ? col : bidx[i];
        }
        __syncthreads();                        // done reading buf p before it is re-staged
    }

    // C layout: VGPR i -> row i (lanes 0-15) / row i+8 (lanes 16-31).
    // Min-reduce over the 16 lanes of each half; ties -> smallest index (argmin semantics).
#pragma unroll
    for (int off = 8; off >= 1; off >>= 1) {
#pragma unroll
        for (int i = 0; i < 8; ++i) {
            float os = __shfl_xor(best[i], off, 32);
            int   oi = __shfl_xor(bidx[i], off, 32);
            bool take = (os < best[i]) || (os == best[i] && oi < bidx[i]);
            best[i] = take ? os : best[i];
            bidx[i] = take ? oi : bidx[i];
        }
    }
    if ((lane & 15) == 0) {
        int rb = rowBase + ((lane >> 4) ? 8 : 0);
#pragma unroll
        for (int i = 0; i < 8; ++i) clusters[rb + i] = bidx[i];
    }
}

// ---------------------------------------------------------------- accumulate sums/counts (one block per row)
__global__ void __launch_bounds__(256)
k_accumulate(const float* __restrict__ x, const int* __restrict__ clusters,
             float* __restrict__ sums, float* __restrict__ counts) {
    int n = blockIdx.x, d = threadIdx.x;
    int c = clusters[n];
    atomicAdd(&sums[(size_t)c * DIM + d], x[(size_t)n * DIM + d]);
    if (d == 0) atomicAdd(&counts[c], 1.0f);
}

// ---------------------------------------------------------------- centroids = sums / counts
__global__ void __launch_bounds__(256)
k_update(const float* __restrict__ sums, const float* __restrict__ counts,
         float* __restrict__ centroids) {
    int k = blockIdx.x, d = threadIdx.x;
    centroids[k * DIM + d] = sums[k * DIM + d] / counts[k];
}

// ---------------------------------------------------------------- pack outputs: clusters(f32), centroids, counts
__global__ void __launch_bounds__(256)
k_finalize(const int* __restrict__ clusters, const float* __restrict__ centroids,
           const float* __restrict__ counts, float* __restrict__ out) {
    int i = blockIdx.x * 256 + threadIdx.x;
    const int total = N_PTS + NCLUST * DIM + NCLUST;
    if (i >= total) return;
    if (i < N_PTS)                     out[i] = (float)clusters[i];
    else if (i < N_PTS + NCLUST * DIM) out[i] = centroids[i - N_PTS];
    else                               out[i] = counts[i - N_PTS - NCLUST * DIM];
}

extern "C" void kernel_launch(void* const* d_in, const int* in_sizes, int n_in,
                              void* d_out, int out_size, void* d_ws, size_t ws_size,
                              hipStream_t stream) {
    const float* x = (const float*)d_in[0];
    float* out = (float*)d_out;

    // Workspace layout (256B-aligned offsets, ~3.6 MB total)
    char* ws = (char*)d_ws;
    float*  centroids = (float*) (ws);                         // 1,048,576 B
    __bf16* chg       = (__bf16*)(ws + 1048576);               //   524,288 B
    __bf16* clg       = (__bf16*)(ws + 1572864);               //   524,288 B
    float*  c2        = (float*) (ws + 2097152);               //     4,096 B
    float*  sums      = (float*) (ws + 2101248);               // 1,048,576 B
    float*  counts    = (float*) (ws + 3149824);               //     4,096 B
    int*    clusters  = (int*)   (ws + 3153920);               //   524,288 B

    // centroids = x[:K]
    k_init_centroids<<<(NCLUST * DIM) / 256, 256, 0, stream>>>(x, centroids);

    for (int it = 0; it < NITERS; ++it) {
        k_prep_centroids<<<NCLUST, 256, 0, stream>>>(centroids, chg, clg, c2);
        k_zero<<<(NCLUST * DIM + NCLUST + 255) / 256, 256, 0, stream>>>(sums, counts);
        k_assign<<<N_PTS / 128, 256, 0, stream>>>(x, chg, clg, c2, clusters);
        k_accumulate<<<N_PTS, 256, 0, stream>>>(x, clusters, sums, counts);
        k_update<<<NCLUST, 256, 0, stream>>>(sums, counts, centroids);
    }
    const int total = N_PTS + NCLUST * DIM + NCLUST;
    k_finalize<<<(total + 255) / 256, 256, 0, stream>>>(clusters, centroids, counts, out);
}
